// ViT_D2ST_Compat_29283087024804
// MI455X (gfx1250) — compile-verified, matrix operand order checked by reading, not run on hardware
//
#include <hip/hip_runtime.h>

// ---------------------------------------------------------------------------
// CDNA5 (gfx1250) deformable-attention pipeline, bf16 WMMA everywhere a
// matmul appears. wave32 WMMA shape 16x16x32 (bf16 in, fp32 accum).
// GEMM v2: weight tile staged in LDS (TDM tensor_load_to_lds when available),
// double-buffered A stream so WMMA overlaps global loads.
// ---------------------------------------------------------------------------

#define DIMX 384
#define BNUM 32
#define TT   8
#define BT   256          // B*T
#define NTOK 197          // HS*WS + 1
#define HSZ  14
#define GC   96           // DIM / GROUPS
#define HC   96           // DIM / HEADS
#define PSAMP 32          // sampled points
#define MTOT (TT * NTOK)  // 1576 query rows per (b,h)

typedef __bf16 bf16_t;
typedef __attribute__((ext_vector_type(16))) __bf16 v16bf;
typedef __attribute__((ext_vector_type(8)))  float  v8f;
typedef __attribute__((ext_vector_type(4)))  unsigned int v4u;
typedef __attribute__((ext_vector_type(4)))  unsigned int u32x4;
typedef __attribute__((ext_vector_type(8)))  int i32x8;
typedef __attribute__((ext_vector_type(4)))  int i32x4;

#if defined(__has_builtin)
#if __has_builtin(__builtin_amdgcn_tensor_load_to_lds) && \
    __has_builtin(__builtin_amdgcn_s_wait_tensorcnt)
#define HAVE_TDM 1
#endif
#endif

union FragBF {
    v16bf v;
    unsigned short h[16];
    v4u q[2];
};

__device__ inline unsigned short f2bf_u16(float f) {
    unsigned int u = __float_as_uint(f);
    u += 0x7FFFu + ((u >> 16) & 1u);          // round-to-nearest-even
    return (unsigned short)(u >> 16);
}
__device__ inline float bf2f_u16(unsigned short s) {
    return __uint_as_float(((unsigned int)s) << 16);
}

__device__ inline v8f v8f_zero() {
    v8f z;
#pragma unroll
    for (int i = 0; i < 8; ++i) z[i] = 0.f;
    return z;
}

// 16x32 bf16 fragment (A-style / transposed-B-style), row-major source.
__device__ inline v16bf load_frag(const unsigned short* __restrict__ p) {
    FragBF f;
    f.q[0] = *(const v4u*)(p);        // K = k0+ksub .. +7
    f.q[1] = *(const v4u*)(p + 16);   // K = k0+ksub+16 .. +23
    return f.v;
}

__device__ inline v8f wmma_bf16(v16bf a, v16bf b, v8f c) {
    return __builtin_amdgcn_wmma_f32_16x16x32_bf16(
        /*neg_a=*/false, a, /*neg_b=*/false, b,
        /*c_mod=*/(short)0, c, /*reuse_a=*/false, /*reuse_b=*/false);
}

// ---------------------------------------------------------------------------
// fp32 -> bf16 conversion (grid-stride)
// ---------------------------------------------------------------------------
__global__ void cvt_kernel(const float* __restrict__ in,
                           unsigned short* __restrict__ out, int n) {
    int i = blockIdx.x * blockDim.x + threadIdx.x;
    int stride = gridDim.x * blockDim.x;
    for (; i < n; i += stride) out[i] = f2bf_u16(in[i]);
}

// ---------------------------------------------------------------------------
// Generic bf16 WMMA GEMM:  C[m,n] = sum_k A[m,k] * W[n,k] + bias[n]
// Block = 256 threads = 8 waves; block tile 128(M) x 64(N); wave tile 16x64.
// The 64 x K bf16 weight tile (48 KB for K=384) is staged in LDS once per
// block -- via the Tensor Data Mover when the builtin exists -- then all
// B-fragments are served from LDS (ds_load_b128). A-fragments are double
// buffered from global so v_wmma overlaps the HBM stream.
// M must be a multiple of 128 (50432 and 1024 both are). K multiple of 32.
// mode 0: row-major store (f32 and/or bf16)
// mode 1: K-layout scatter  kbuf[((b*4+h)*32+p)*96+c]
// mode 2: V-layout scatter  vbuf[((b*4+h)*96+c)*32+p]
// ---------------------------------------------------------------------------
__global__ __launch_bounds__(256) void gemm_wmma(
    const unsigned short* __restrict__ A, const unsigned short* __restrict__ W,
    const float* __restrict__ bias, float* __restrict__ Cf,
    unsigned short* __restrict__ Cb, int M, int Ncol, int K, int mode) {
    __shared__ unsigned short ldsW[64 * DIMX];   // 48 KB (K <= DIMX assumed)

    const int wave = threadIdx.x >> 5;
    const int lane = threadIdx.x & 31;
    const int row0 = blockIdx.y * 128 + wave * 16;
    const int col0 = blockIdx.x * 64;
    const int r = lane & 15;
    const int ksub = (lane >> 4) << 3;     // 0 for lanes 0-15, 8 for 16-31

    // ---- stage 64 consecutive weight rows (contiguous 64*K halves) in LDS
    const unsigned short* wbase = W + (size_t)col0 * K;
#ifdef HAVE_TDM
    if (wave == 0) {
        const unsigned long long ga = (unsigned long long)(uintptr_t)wbase;
        const unsigned int ldsoff =
            (unsigned int)(uintptr_t)(&ldsW[0]);   // LDS byte offset
        u32x4 g0;
        g0[0] = 0x1u;                               // count=1, user descriptor
        g0[1] = ldsoff;                             // lds_addr
        g0[2] = (unsigned int)ga;                   // global_addr[31:0]
        g0[3] = ((unsigned int)(ga >> 32) & 0x01FFFFFFu) |
                0x80000000u;                        // addr[56:32] | type=2
        i32x8 g1;
        g1[0] = 0x00010000;                         // data_size = 2 bytes
        g1[1] = (int)((unsigned)(K & 0xFFFF) << 16);        // tensor_dim0 lo
        g1[2] = (int)(((unsigned)K >> 16) | (64u << 16));   // dim0 hi|dim1 lo
        g1[3] = (int)((unsigned)(K & 0xFFFF) << 16);        // dim1 hi|tile_dim0
        g1[4] = 64;                                 // tile_dim1 (tile_dim2=0)
        g1[5] = K;                                  // tensor_dim0_stride lo
        g1[6] = 0;
        g1[7] = 0;
        i32x4 z4;
        z4[0] = 0; z4[1] = 0; z4[2] = 0; z4[3] = 0;
#if __clang_major__ >= 23
        i32x8 z8;
#pragma unroll
        for (int i = 0; i < 8; ++i) z8[i] = 0;
        __builtin_amdgcn_tensor_load_to_lds(g0, g1, z4, z4, z8, 0);
#else
        __builtin_amdgcn_tensor_load_to_lds(g0, g1, z4, z4, 0);
#endif
        __builtin_amdgcn_s_wait_tensorcnt((short)0);
    }
#else
    {
        const int chunks = 8 * K;                   // (64*K)/8 16-byte chunks
        for (int i = threadIdx.x; i < chunks; i += 256)
            ((v4u*)ldsW)[i] = ((const v4u*)wbase)[i];
    }
#endif
    __syncthreads();

    const unsigned short* arow = A + (size_t)(row0 + r) * K + ksub;
    const unsigned short* lrow = ldsW + (size_t)r * K + ksub;

    v8f acc[4];
#pragma unroll
    for (int j = 0; j < 4; ++j) acc[j] = v8f_zero();

    // ---- double-buffered K loop
    FragBF a_cur;
    a_cur.q[0] = *(const v4u*)(arow);
    a_cur.q[1] = *(const v4u*)(arow + 16);
    for (int k0 = 0; k0 < K; k0 += 32) {
        const int kn = (k0 + 32 < K) ? (k0 + 32) : k0;   // clamp last prefetch
        FragBF a_nxt;
        a_nxt.q[0] = *(const v4u*)(arow + kn);
        a_nxt.q[1] = *(const v4u*)(arow + kn + 16);
        __builtin_prefetch((const void*)(arow + kn + 32), 0, 1);
        const unsigned short* lp = lrow + k0;
#pragma unroll
        for (int j = 0; j < 4; ++j) {
            FragBF b;
            b.q[0] = *(const v4u*)(lp);
            b.q[1] = *(const v4u*)(lp + 16);
            lp += 16 * K;
            acc[j] = wmma_bf16(a_cur.v, b.v, acc[j]);
        }
        a_cur = a_nxt;
    }

    // ---- epilogue
    const int mb = row0 + ((lane & 16) ? 8 : 0);
#pragma unroll
    for (int j = 0; j < 4; ++j) {
        const int nIdx = col0 + j * 16 + r;
        const float bvv = bias ? bias[nIdx] : 0.f;
#pragma unroll
        for (int i = 0; i < 8; ++i) {
            const int mm = mb + i;
            const float val = acc[j][i] + bvv;
            if (mode == 0) {
                if (Cf) Cf[(size_t)mm * Ncol + nIdx] = val;
                if (Cb) Cb[(size_t)mm * Ncol + nIdx] = f2bf_u16(val);
            } else {
                const int hh = nIdx / HC, cc = nIdx - hh * HC;
                const int bb = mm >> 5, pp = mm & 31;
                if (mode == 1)
                    Cb[((size_t)((bb * 4 + hh) * PSAMP) + pp) * HC + cc] =
                        f2bf_u16(val);
                else
                    Cb[((size_t)((bb * 4 + hh) * HC) + cc) * PSAMP + pp] =
                        f2bf_u16(val);
            }
        }
    }
}

// ---------------------------------------------------------------------------
// Offset head: depthwise conv3d(4x5x5, stride 4,3,3) + channel-LN + GELU +
// 1x1x1 conv(96->3) + tanh*range + reference points -> sampling grid (xyz).
// ---------------------------------------------------------------------------
__global__ __launch_bounds__(128) void offsets_kernel(
    const unsigned short* __restrict__ qb, const float* __restrict__ w1,
    const float* __restrict__ b1, const float* __restrict__ lng,
    const float* __restrict__ lnb, const float* __restrict__ w2,
    float* __restrict__ grid) {
    const int blk = blockIdx.x;               // bg*32 + pos
    const int bg = blk >> 5, pos = blk & 31;
    const int b = bg >> 2, g = bg & 3;
    const int tp = pos >> 4, hp = (pos >> 2) & 3, wp = pos & 3;
    const int c = threadIdx.x;

    __shared__ float sconv[GC];
    __shared__ float sg[GC];
    __shared__ float sred[3];

    if (c < GC) {
        float acc = 0.f;
        const float* wc = w1 + c * 100;
        const int d = g * GC + c;
#pragma unroll
        for (int kt = 0; kt < 4; ++kt) {
            const int t = tp * 4 + kt;
            const int bt = b * TT + t;
            for (int kh = 0; kh < 5; ++kh) {
                const int hh = hp * 3 + kh;
                for (int kw = 0; kw < 5; ++kw) {
                    const int ww = wp * 3 + kw;
                    const int tok = 1 + hh * HSZ + ww;
                    const float v =
                        bf2f_u16(qb[((size_t)tok * BT + bt) * DIMX + d]);
                    acc += v * wc[kt * 25 + kh * 5 + kw];
                }
            }
        }
        sconv[c] = acc + b1[c];
    }
    __syncthreads();

    float mu = 0.f, m2 = 0.f;
    for (int i = 0; i < GC; ++i) {
        const float v = sconv[i];
        mu += v;
        m2 += v * v;
    }
    mu *= (1.f / GC);
    m2 = m2 * (1.f / GC) - mu * mu;        // biased variance
    const float inv = rsqrtf(m2 + 1e-5f);

    if (c < GC) {
        const float v = (sconv[c] - mu) * inv * lng[c] + lnb[c];
        sg[c] = 0.5f * v * (1.f + erff(v * 0.70710678118654752f)); // exact gelu
    }
    __syncthreads();

    if (c < 3) {
        float s = 0.f;
        for (int i = 0; i < GC; ++i) s += w2[c * GC + i] * sg[i];
        sred[c] = s;
    }
    __syncthreads();

    if (c == 0) {
        const float offz = tanhf(sred[0]) * 1.0f;  // min(1, 2/TP) = 1
        const float offy = tanhf(sred[1]) * 0.5f;  // min(1, 2/HP) = .5
        const float offx = tanhf(sred[2]) * 0.5f;
        const float rz = (float)tp - 0.5f;
        const float ry = ((float)hp + 0.5f) * 0.5f - 1.f;
        const float rx = ((float)wp + 0.5f) * 0.5f - 1.f;
        float* gp = grid + ((size_t)bg * PSAMP + pos) * 3;
        gp[0] = offx + rx;   // zyx -> xyz reversal
        gp[1] = offy + ry;
        gp[2] = offz + rz;
    }
}

// ---------------------------------------------------------------------------
// Trilinear grid_sample (align_corners=True, zeros padding) of patch tokens.
// ---------------------------------------------------------------------------
__global__ __launch_bounds__(128) void sample_kernel(
    const float* __restrict__ x, const float* __restrict__ grid,
    unsigned short* __restrict__ xsmp) {
    const int blk = blockIdx.x;
    const int bg = blk >> 5, pos = blk & 31;
    const int b = bg >> 2, g = bg & 3;
    const int c = threadIdx.x;
    if (c >= GC) return;

    const float* gp = grid + ((size_t)bg * PSAMP + pos) * 3;
    const float ix = (gp[0] + 1.f) * 0.5f * (HSZ - 1);
    const float iy = (gp[1] + 1.f) * 0.5f * (HSZ - 1);
    const float iz = (gp[2] + 1.f) * 0.5f * (TT - 1);
    const float x0f = floorf(ix), y0f = floorf(iy), z0f = floorf(iz);
    const float wx1 = ix - x0f, wy1 = iy - y0f, wz1 = iz - z0f;
    const int x0 = (int)x0f, y0 = (int)y0f, z0 = (int)z0f;
    const int d = g * GC + c;

    float acc = 0.f;
#pragma unroll
    for (int dz = 0; dz < 2; ++dz) {
        const int zc = z0 + dz;
        const float wz = dz ? wz1 : 1.f - wz1;
#pragma unroll
        for (int dy = 0; dy < 2; ++dy) {
            const int yc = y0 + dy;
            const float wy = dy ? wy1 : 1.f - wy1;
#pragma unroll
            for (int dx = 0; dx < 2; ++dx) {
                const int xc = x0 + dx;
                const float wx = dx ? wx1 : 1.f - wx1;
                const bool valid = (zc >= 0 && zc < TT && yc >= 0 &&
                                    yc < HSZ && xc >= 0 && xc < HSZ);
                const int zcc = min(max(zc, 0), TT - 1);
                const int ycc = min(max(yc, 0), HSZ - 1);
                const int xcc = min(max(xc, 0), HSZ - 1);
                const int tok = 1 + ycc * HSZ + xcc;
                const int bt = b * TT + zcc;
                const float v = x[((size_t)tok * BT + bt) * DIMX + d];
                acc += v * (valid ? (wz * wy * wx) : 0.f);
            }
        }
    }
    xsmp[((size_t)(b * PSAMP + pos)) * DIMX + d] = f2bf_u16(acc);
}

// ---------------------------------------------------------------------------
// Attention: per (b,h) all 1576 query tokens attend to 32 sampled keys.
// scores (WMMA, K=96) -> softmax (16-lane shfl reductions, wave32) ->
// LDS re-swizzle to A-fragment -> out = P @ V (WMMA, K=32) -> bf16 scatter.
// ---------------------------------------------------------------------------
__global__ __launch_bounds__(256) void attn_kernel(
    const unsigned short* __restrict__ qb,
    const unsigned short* __restrict__ kbuf,
    const unsigned short* __restrict__ vbuf,
    unsigned short* __restrict__ attnout) {
    const int bh = blockIdx.x;                // b*4 + h
    const int b = bh >> 2, h = bh & 3;
    const int wave = threadIdx.x >> 5;
    const int lane = threadIdx.x & 31;
    const int mtile = blockIdx.y * 128 + wave * 16;
    const int r = lane & 15;
    const int ksub = (lane >> 4) << 3;

    __shared__ float probs[8][16][33];        // per-wave 16x32 prob tiles

    int m = mtile + r;
    if (m > MTOT - 1) m = MTOT - 1;
    const int t = m / NTOK;
    const int n = m - t * NTOK;
    const unsigned short* qrow =
        qb + ((size_t)n * BT + (b * TT + t)) * DIMX + h * HC;

    const unsigned short* kb = kbuf + (size_t)bh * PSAMP * HC;
    const unsigned short* vb = vbuf + (size_t)bh * HC * PSAMP;

    v8f acc0 = v8f_zero();
    v8f acc1 = v8f_zero();
#pragma unroll
    for (int c0 = 0; c0 < HC; c0 += 32) {
        const v16bf a = load_frag(qrow + c0 + ksub);
        const v16bf b0 = load_frag(kb + (size_t)r * HC + c0 + ksub);
        const v16bf b1 = load_frag(kb + (size_t)(16 + r) * HC + c0 + ksub);
        acc0 = wmma_bf16(a, b0, acc0);
        acc1 = wmma_bf16(a, b1, acc1);
    }

    const float scale = 0.1020620726159657f;  // HC^-0.5
#pragma unroll
    for (int rr = 0; rr < 8; ++rr) {
        float s0 = acc0[rr] * scale;
        float s1 = acc1[rr] * scale;
        float mx = fmaxf(s0, s1);
        mx = fmaxf(mx, __shfl_xor(mx, 1, 16));
        mx = fmaxf(mx, __shfl_xor(mx, 2, 16));
        mx = fmaxf(mx, __shfl_xor(mx, 4, 16));
        mx = fmaxf(mx, __shfl_xor(mx, 8, 16));
        float e0 = __expf(s0 - mx);
        float e1 = __expf(s1 - mx);
        float sm = e0 + e1;
        sm += __shfl_xor(sm, 1, 16);
        sm += __shfl_xor(sm, 2, 16);
        sm += __shfl_xor(sm, 4, 16);
        sm += __shfl_xor(sm, 8, 16);
        const float inv = 1.f / sm;
        const int row = rr + ((lane & 16) ? 8 : 0);
        probs[wave][row][r] = e0 * inv;
        probs[wave][row][r + 16] = e1 * inv;
    }
    __syncthreads();

    FragBF a2;
#pragma unroll
    for (int i = 0; i < 8; ++i)
        a2.h[i] = f2bf_u16(probs[wave][r][ksub + i]);
#pragma unroll
    for (int i = 0; i < 8; ++i)
        a2.h[8 + i] = f2bf_u16(probs[wave][r][16 + ksub + i]);

    const int mb = mtile + ((lane & 16) ? 8 : 0);
#pragma unroll
    for (int j = 0; j < 6; ++j) {             // 6 column tiles cover HC=96
        const v16bf bvf = load_frag(vb + (size_t)(j * 16 + r) * PSAMP + ksub);
        v8f o = wmma_bf16(a2.v, bvf, v8f_zero());
        const int cc = h * HC + j * 16 + r;
#pragma unroll
        for (int i = 0; i < 8; ++i) {
            const int mm = mb + i;
            if (mm < MTOT) {
                const int tt = mm / NTOK;
                const int nn = mm - tt * NTOK;
                attnout[((size_t)nn * BT + b * TT + tt) * DIMX + cc] =
                    f2bf_u16(o[i]);
            }
        }
    }
}

// ---------------------------------------------------------------------------
// Host launch
// ---------------------------------------------------------------------------
extern "C" void kernel_launch(void* const* d_in, const int* in_sizes, int n_in,
                              void* d_out, int out_size, void* d_ws,
                              size_t ws_size, hipStream_t stream) {
    (void)in_sizes; (void)n_in; (void)out_size; (void)ws_size;
    const float* x   = (const float*)d_in[0];
    const float* wq  = (const float*)d_in[1];
    const float* bq  = (const float*)d_in[2];
    const float* wk  = (const float*)d_in[3];
    const float* bk  = (const float*)d_in[4];
    const float* wv  = (const float*)d_in[5];
    const float* bv  = (const float*)d_in[6];
    const float* wo  = (const float*)d_in[7];
    const float* bo  = (const float*)d_in[8];
    const float* c1w = (const float*)d_in[9];
    const float* c1b = (const float*)d_in[10];
    const float* lng = (const float*)d_in[11];
    const float* lnb = (const float*)d_in[12];
    const float* c2w = (const float*)d_in[13];

    const size_t NX = (size_t)NTOK * BT * DIMX;   // 19,365,888 elements
    const size_t NW = (size_t)DIMX * DIMX;        // 147,456 elements

    char* base = (char*)d_ws;
    size_t off = 0;
    auto take = [&](size_t bytes) -> char* {
        char* p = base + off;
        off = (off + bytes + 255) & ~(size_t)255;
        return p;
    };
    unsigned short* xbf  = (unsigned short*)take(NX * 2);
    unsigned short* qbf  = (unsigned short*)take(NX * 2);
    unsigned short* aobf = (unsigned short*)take(NX * 2);
    unsigned short* wqbf = (unsigned short*)take(NW * 2);
    unsigned short* wkbf = (unsigned short*)take(NW * 2);
    unsigned short* wvbf = (unsigned short*)take(NW * 2);
    unsigned short* wobf = (unsigned short*)take(NW * 2);
    unsigned short* xsbf = (unsigned short*)take((size_t)BNUM * PSAMP * DIMX * 2);
    unsigned short* kbuf = (unsigned short*)take((size_t)128 * PSAMP * HC * 2);
    unsigned short* vbuf = (unsigned short*)take((size_t)128 * HC * PSAMP * 2);
    float* gridbuf       = (float*)take((size_t)128 * PSAMP * 3 * 4);

    const int M_BIG = NTOK * BT;    // 50432 = 394 * 128
    const int M_KV  = BNUM * PSAMP; // 1024 = 8 * 128

    // 1) bf16 conversions
    cvt_kernel<<<4096, 256, 0, stream>>>(x, xbf, (int)NX);
    cvt_kernel<<<576, 256, 0, stream>>>(wq, wqbf, (int)NW);
    cvt_kernel<<<576, 256, 0, stream>>>(wk, wkbf, (int)NW);
    cvt_kernel<<<576, 256, 0, stream>>>(wv, wvbf, (int)NW);
    cvt_kernel<<<576, 256, 0, stream>>>(wo, wobf, (int)NW);

    // 2) q = x @ wq^T + bq  (bf16 out)
    gemm_wmma<<<dim3(DIMX / 64, M_BIG / 128), 256, 0, stream>>>(
        xbf, wqbf, bq, nullptr, qbf, M_BIG, DIMX, DIMX, 0);

    // 3) offset head -> sampling grid
    offsets_kernel<<<4096, 128, 0, stream>>>(qbf, c1w, c1b, lng, lnb, c2w,
                                             gridbuf);

    // 4) trilinear sampling of patch tokens
    sample_kernel<<<4096, 128, 0, stream>>>(x, gridbuf, xsbf);

    // 5) K/V projections with attention-friendly layouts
    gemm_wmma<<<dim3(DIMX / 64, M_KV / 128), 256, 0, stream>>>(
        xsbf, wkbf, bk, nullptr, kbuf, M_KV, DIMX, DIMX, 1);
    gemm_wmma<<<dim3(DIMX / 64, M_KV / 128), 256, 0, stream>>>(
        xsbf, wvbf, bv, nullptr, vbuf, M_KV, DIMX, DIMX, 2);

    // 6) attention (128 (b,h) pairs x 13 row-tiles of 128)
    attn_kernel<<<dim3(128, (MTOT + 127) / 128), 256, 0, stream>>>(
        qbf, kbuf, vbuf, aobf);

    // 7) final projection -> fp32 output
    gemm_wmma<<<dim3(DIMX / 64, M_BIG / 128), 256, 0, stream>>>(
        aobf, wobf, bo, (float*)d_out, nullptr, M_BIG, DIMX, DIMX, 0);
}
